// LRPredictor_62474594288157
// MI455X (gfx1250) — matrix operand order repaired and checked by reading, not compiled, not run on hardware
//
#include <hip/hip_runtime.h>
#include <hip/hip_bf16.h>
#include <stdint.h>

// CDNA5 / gfx1250, wave32. B=262144, T=100, H=32, gates=96, MLP hidden=64.

typedef _Float16 v16h __attribute__((ext_vector_type(16)));
typedef float    v8f  __attribute__((ext_vector_type(8)));

#define GATE_N   96
#define MLP_M    64
#define LR_MIN_C 1e-08f
#define LR_MAX_C 100.0f

static __device__ __forceinline__ uint32_t pkh(float a, float b) {
  auto p = __builtin_amdgcn_cvt_pkrtz(a, b);   // v_cvt_pk_rtz_f16_f32: a->[15:0], b->[31:16]
  return __builtin_bit_cast(uint32_t, p);
}

// SWAPX16: group-of-32 ds_swizzle, xor_mask=0x10, and_mask=0x1f -> offset 0x401F
static __device__ __forceinline__ uint32_t swap16u(uint32_t x) {
  return (uint32_t)__builtin_amdgcn_ds_swizzle((int)x, 0x401F);
}
static __device__ __forceinline__ float swap16f(float x) {
  return __builtin_bit_cast(float, swap16u(__builtin_bit_cast(uint32_t, x)));
}

static __device__ __forceinline__ v8f wmma_f16(v16h a, v16h b, v8f c) {
  // v_wmma_f32_16x16x32_f16  D = A(16x32,f16) * B(32x16,f16) + C(16x16,f32)
  return __builtin_amdgcn_wmma_f32_16x16x32_f16(false, a, false, b, (short)0, c, false, false);
}

static __device__ __forceinline__ float fsig(float x) {
  float e = __builtin_amdgcn_exp2f(x * -1.4426950408889634f);   // exp(-x)
  return __builtin_amdgcn_rcpf(1.0f + e);
}
static __device__ __forceinline__ float ftanh(float x) {
  float e = __builtin_amdgcn_exp2f(x * -2.8853900817779268f);   // exp(-2x)
  return (1.0f - e) * __builtin_amdgcn_rcpf(1.0f + e);
}

// Async global->LDS copies (CDNA5 VGLOBAL async ops, tracked with ASYNCcnt).
// GVS addressing: mem = SGPR_base + VGPR_i32_offset; vdst = LDS byte address.
static __device__ __forceinline__ void async_g2l_b128(uint32_t lds_byte,
                                                      uint32_t gbyte_off,
                                                      const float* gbase) {
  asm volatile("global_load_async_to_lds_b128 %0, %1, %2"
               :: "v"(lds_byte), "v"(gbyte_off), "s"(gbase) : "memory");
}
static __device__ __forceinline__ void async_g2l_b32(uint32_t lds_byte,
                                                     uint32_t gbyte_off,
                                                     const float* gbase) {
  asm volatile("global_load_async_to_lds_b32 %0, %1, %2"
               :: "v"(lds_byte), "v"(gbyte_off), "s"(gbase) : "memory");
}
static __device__ __forceinline__ void wait_async0() {
  asm volatile("s_wait_asynccnt 0x0" ::: "memory");
}

// 16x32 f16 A-tile of a transposed weight: A[m][k] = W[(koff+k)*ld + gbase+m]
// Layout (contiguous K, matching buildB): lanes 0-15 row m hold K=0..15,
// lanes 16-31 row m hold K=16..31; 2 halfs per VGPR, K-even in [15:0].
static __device__ __forceinline__ v16h loadA(const float* __restrict__ W, int ld,
                                             int gbase, int koff, int lane) {
  const int m  = lane & 15;
  const int kb = (lane & 16) ? 16 : 0;
  union { v16h v; uint32_t u[8]; } r;
#pragma unroll
  for (int s = 0; s < 8; ++s) {
    int k0 = kb + 2 * s;
    r.u[s] = pkh(W[(koff + k0) * ld + gbase + m],
                 W[(koff + k0 + 1) * ld + gbase + m]);
  }
  return r.v;
}

// Convert a 32x16 f32 matrix held as two 16x16 C/D-layout tiles (t0 = rows 0-15,
// t1 = rows 16-31; VGPR r: lanes 0-15 = row r, lanes 16-31 = row r+8) into the
// contiguous-K f16 B-operand layout this code pairs with loadA: VGPR v lanes 0-15 = K{2v,2v+1},
// lanes 16-31 = K{16+2v,17+2v}. Needs only a lane^16 swap (ds_swizzle SWAPX16).
static __device__ __forceinline__ v16h buildB(v8f t0, v8f t1, int lane) {
  const bool lo = (lane & 16) == 0;
  union { v16h v; uint32_t u[8]; } r;
#pragma unroll
  for (int i = 0; i < 4; ++i) {                   // K rows 0-7 / 16-23
    uint32_t a  = pkh(t0[2*i], t0[2*i+1]);        // rows 2i,2i+1   valid in lanes 0-15
    uint32_t b  = pkh(t1[2*i], t1[2*i+1]);        // rows 16+2i,... valid in lanes 0-15
    uint32_t bs = swap16u(b);                     // move to lanes 16-31 (convergent)
    r.u[i] = lo ? a : bs;
  }
#pragma unroll
  for (int i = 4; i < 8; ++i) {                   // K rows 8-15 / 24-31
    uint32_t a  = pkh(t0[2*i-8], t0[2*i-7]);      // rows 2i,2i+1   valid in lanes 16-31
    uint32_t b  = pkh(t1[2*i-8], t1[2*i-7]);      // rows 16+2i     valid in lanes 16-31
    uint32_t as = swap16u(a);                     // move to lanes 0-15
    r.u[i] = lo ? as : b;
  }
  return r.v;
}

__global__ __launch_bounds__(256, 1)
void lr_pred_kernel(const float* __restrict__ lh, const float* __restrict__ wn,
                    const float* __restrict__ gn, const float* __restrict__ lc,
                    const float* __restrict__ sp, const float* __restrict__ Wi,
                    const float* __restrict__ Wh, const float* __restrict__ bh,
                    const float* __restrict__ W1, const float* __restrict__ b1,
                    const float* __restrict__ W2, const float* __restrict__ b2,
                    const float* __restrict__ W3, const float* __restrict__ b3,
                    float* __restrict__ out, int Bn, int T) {
  extern __shared__ float smem[];               // waves_per_block * 16 * T floats
  const int lane = (int)(threadIdx.x & 31u);
  const int widx = (int)(threadIdx.x >> 5);     // wave within block
  const int wave = (int)((blockIdx.x * blockDim.x + threadIdx.x) >> 5);
  const int base = wave * 16;                   // 16 batch columns per wave
  if (base >= Bn) return;                       // wave-uniform guard (EXEC stays full)

  // ---- Stage this wave's loss_history tile (16 contiguous rows = 16*T floats,
  // contiguous in global!) into LDS via the async global->LDS path.
  float* wtile = smem + (size_t)widx * 16 * T;
  {
    const float* gsrc = lh + (size_t)base * (size_t)T;
    const uint32_t lds0 = (uint32_t)(uintptr_t)wtile;  // low 32 bits = LDS offset
    const int totalF = 16 * T;
    const int vecEnd = totalF & ~127;           // 32 lanes * 4 floats per pass
    for (int i = lane * 4; i < vecEnd; i += 128)
      async_g2l_b128(lds0 + (uint32_t)i * 4u, (uint32_t)i * 4u, gsrc);
    for (int i = vecEnd + lane; i < totalF; i += 32)
      async_g2l_b32(lds0 + (uint32_t)i * 4u, (uint32_t)i * 4u, gsrc);
  }

  // Wh^T WMMA A-tiles (6 x 16x32 f16), register-resident across all 100 steps.
  v16h Awh[6];
#pragma unroll
  for (int t = 0; t < 6; ++t) Awh[t] = loadA(Wh, GATE_N, 16 * t, 0, lane);

  // Per-lane gate constants: cell (q,s) covers gate-row 16q + s + 8*(lane>=16).
  float wiR[16], wiZ[16], wiN[16], bhR[16], bhZ[16], bhN[16];
#pragma unroll
  for (int q = 0; q < 2; ++q)
#pragma unroll
    for (int s = 0; s < 8; ++s) {
      int cell = q * 8 + s;
      int row  = 16 * q + s + ((lane & 16) ? 8 : 0);
      wiR[cell] = Wi[row];  wiZ[cell] = Wi[32 + row];  wiN[cell] = Wi[64 + row];
      bhR[cell] = bh[row];  bhZ[cell] = bh[32 + row];  bhN[cell] = bh[64 + row];
    }

  const int bcol = base + (lane & 15);
  const float* xlds = wtile + (size_t)(lane & 15) * T;  // LDS row, conflict-free banks

  // GRU state h^T (32 x 16) as two 16x16 f32 accumulator-layout tiles, zero init.
  const v8f zero = {0.f, 0.f, 0.f, 0.f, 0.f, 0.f, 0.f, 0.f};
  v8f h0 = zero, h1 = zero;

  wait_async0();                                // tile resident in LDS from here on

  for (int t = 0; t < T; ++t) {
    v16h Bh = buildB(h0, h1, lane);
    // gh^T = Wh^T(96x32) @ h^T(32x16): 6 WMMAs, K=32 == H in one shot.
    v8f aR0 = wmma_f16(Awh[0], Bh, zero);   // r rows  0-15
    v8f aR1 = wmma_f16(Awh[1], Bh, zero);   // r rows 16-31
    v8f aZ0 = wmma_f16(Awh[2], Bh, zero);   // z rows  0-15
    v8f aZ1 = wmma_f16(Awh[3], Bh, zero);   // z rows 16-31
    v8f aN0 = wmma_f16(Awh[4], Bh, zero);   // n rows  0-15
    v8f aN1 = wmma_f16(Awh[5], Bh, zero);   // n rows 16-31

    float x = xlds[t];                      // ds_load_b32, 16 distinct banks
#pragma unroll
    for (int s = 0; s < 8; ++s) {
      // tile q=0 (h rows 0-15)
      {
        float r   = fsig(fmaf(x, wiR[s], aR0[s] + bhR[s]));
        float z   = fsig(fmaf(x, wiZ[s], aZ0[s] + bhZ[s]));
        float ghn = aN0[s] + bhN[s];
        float n   = ftanh(fmaf(r, ghn, x * wiN[s]));
        h0[s] = fmaf(z, h0[s] - n, n);        // (1-z)*n + z*h
      }
      // tile q=1 (h rows 16-31)
      {
        float r   = fsig(fmaf(x, wiR[8+s], aR1[s] + bhR[8+s]));
        float z   = fsig(fmaf(x, wiZ[8+s], aZ1[s] + bhZ[8+s]));
        float ghn = aN1[s] + bhN[8+s];
        float n   = ftanh(fmaf(r, ghn, x * wiN[8+s]));
        h1[s] = fmaf(z, h1[s] - n, n);
      }
    }
  }

  // ---------------- MLP head (transposed) ----------------
  // Layer 1: a1^T = relu(W1[4:36]^T @ h^T + rank-4 scalar part + b1)
  v16h Bh = buildB(h0, h1, lane);
  v8f a1[4];
#pragma unroll
  for (int t = 0; t < 4; ++t) {
    v16h A1 = loadA(W1, MLP_M, 16 * t, /*koff=*/4, lane);   // skip 4 scalar rows
    a1[t] = wmma_f16(A1, Bh, zero);
  }
  float s0 = wn[bcol], s1 = gn[bcol], s2 = lc[bcol], s3 = sp[bcol];
#pragma unroll
  for (int t = 0; t < 4; ++t)
#pragma unroll
    for (int s = 0; s < 8; ++s) {
      int m = 16 * t + s + ((lane & 16) ? 8 : 0);
      float v = a1[t][s] + b1[m];
      v = fmaf(s0, W1[m],        v);    // W1[0][m]
      v = fmaf(s1, W1[64  + m],  v);    // W1[1][m]
      v = fmaf(s2, W1[128 + m],  v);    // W1[2][m]
      v = fmaf(s3, W1[192 + m],  v);    // W1[3][m]
      a1[t][s] = fmaxf(v, 0.f);
    }

  // Layer 2: a2^T = relu(W2^T(64x64) @ a1^T), K=64 -> two K=32 chunks.
  v16h B0 = buildB(a1[0], a1[1], lane);   // K rows  0-31
  v16h B1 = buildB(a1[2], a1[3], lane);   // K rows 32-63
  v8f a2[4];
#pragma unroll
  for (int t = 0; t < 4; ++t) {
    v16h A2a = loadA(W2, MLP_M, 16 * t, 0,  lane);
    v16h A2b = loadA(W2, MLP_M, 16 * t, 32, lane);
    a2[t] = wmma_f16(A2b, B1, wmma_f16(A2a, B0, zero));
  }

  // Layer 3: logit[b] = sum_m relu(a2+b2)[m,b] * W3[m]  (+ b3)
  float part = 0.f;
#pragma unroll
  for (int t = 0; t < 4; ++t)
#pragma unroll
    for (int s = 0; s < 8; ++s) {
      int m = 16 * t + s + ((lane & 16) ? 8 : 0);
      float v = fmaxf(a2[t][s] + b2[m], 0.f);
      part = fmaf(v, W3[m], part);
    }
  part += swap16f(part);                 // combine lane-half partials (all lanes exec)

  if (lane < 16) {
    float logit = part + b3[0];
    out[bcol] = fsig(logit) * (LR_MAX_C - LR_MIN_C) + LR_MIN_C;
  }
}

extern "C" void kernel_launch(void* const* d_in, const int* in_sizes, int n_in,
                              void* d_out, int out_size, void* d_ws, size_t ws_size,
                              hipStream_t stream) {
  (void)n_in; (void)d_ws; (void)ws_size; (void)out_size;
  const float* lh = (const float*)d_in[0];
  const float* wn = (const float*)d_in[1];
  const float* gn = (const float*)d_in[2];
  const float* lc = (const float*)d_in[3];
  const float* sp = (const float*)d_in[4];
  const float* Wi = (const float*)d_in[5];
  const float* Wh = (const float*)d_in[6];
  const float* bh = (const float*)d_in[7];
  const float* W1 = (const float*)d_in[8];
  const float* b1 = (const float*)d_in[9];
  const float* W2 = (const float*)d_in[10];
  const float* b2 = (const float*)d_in[11];
  const float* W3 = (const float*)d_in[12];
  const float* b3 = (const float*)d_in[13];
  float* out = (float*)d_out;

  const int Bn = in_sizes[1];                 // weight_norm is [B]
  const int T  = in_sizes[0] / Bn;            // loss_history is [B,T]

  const int THREADS = 256;                    // 8 waves/block, 16 batch cols/wave
  const int waves   = (Bn + 15) / 16;
  const int wpb     = THREADS / 32;
  const int blocks  = (waves + wpb - 1) / wpb;
  const size_t shmem = (size_t)wpb * 16 * T * sizeof(float);  // 50 KB @ T=100

  lr_pred_kernel<<<blocks, THREADS, shmem, stream>>>(
      lh, wn, gn, lc, sp, Wi, Wh, bh, W1, b1, W2, b2, W3, b3, out, Bn, T);
}
